// MAB_2121713844542
// MI455X (gfx1250) — matrix operand order repaired and checked by reading, not compiled, not run on hardware
//
#include <hip/hip_runtime.h>
#include <math.h>

// ---------------- problem constants ----------------
#define BB   32
#define SS   512
#define DD   256      // DQ = DK = DV
#define HH   8
#define DHD  32       // DV / H
#define DFF  2048
#define NROW (BB*SS)  // 16384

// ---------------- WMMA vector types ----------------
typedef __attribute__((ext_vector_type(16))) __bf16 bf16x16;
typedef __attribute__((ext_vector_type(8)))  float  floatx8;

union ABPack { bf16x16 v; unsigned int u[8]; };

__device__ __forceinline__ unsigned short f2bf_bits(float f) {
    union { float f; unsigned u; } c; c.f = f;
    unsigned r = c.u + 0x7FFFu + ((c.u >> 16) & 1u);   // round-to-nearest-even
    return (unsigned short)(r >> 16);
}
__device__ __forceinline__ unsigned int pack_bf2(float a, float b) {
    return (unsigned int)f2bf_bits(a) | ((unsigned int)f2bf_bits(b) << 16);
}
__device__ __forceinline__ float bf2f(unsigned short u) {
    union { unsigned u; float f; } c; c.u = ((unsigned)u) << 16;
    return c.f;
}
__device__ __forceinline__ float gelu_exact(float x) {
    return 0.5f * x * (1.0f + erff(x * 0.70710678118654752f));
}

// ---------------- weight convert + transpose: WT[n][k] = bf16(W[k][n]) ----------------
__global__ void tconv_kernel(const float* __restrict__ W, unsigned short* __restrict__ WT,
                             int K, int N) {
    long i = (long)blockIdx.x * 256 + threadIdx.x;
    if (i < (long)K * N) {
        int n = (int)(i % N);
        int k = (int)(i / N);
        WT[(long)n * K + k] = f2bf_bits(W[(long)k * N + n]);
    }
}

// ---------------- positional projection Ph = p @ Wp + bp (K=3, pure VALU) ----------------
__global__ void pe_kernel(const float* __restrict__ p, const float* __restrict__ Wp,
                          const float* __restrict__ bp, unsigned short* __restrict__ Ph) {
    int row = blockIdx.x;           // 0 .. NROW-1
    int t   = threadIdx.x;          // 0 .. 255
    float p0 = p[row * 3 + 0], p1 = p[row * 3 + 1], p2 = p[row * 3 + 2];
    float v = p0 * Wp[t] + p1 * Wp[256 + t] + p2 * Wp[512 + t] + bp[t];
    Ph[(long)row * DD + t] = f2bf_bits(v);
}

// ---------------- WMMA GEMM: C = A[M,K] * B[K,N] + bias ----------------
// Block = 128 threads (4 waves). Each wave owns a 16x64 output strip
// (4 accumulators); each main-loop iteration consumes K-chunk of 64
// => 8 v_wmma per wave per barrier pair, A regs reused 4x.
// A: fp32 row-major (ABF16=false) or bf16 row-major (ABF16=true)
// BT: bf16 transposed [N][K]
// EPI: 0 = store bf16 ; 1 = gelu -> bf16 ; 2 = store fp32
template<bool ABF16, int EPI>
__global__ __launch_bounds__(128) void gemm_kernel(
    const void* __restrict__ Av, const unsigned short* __restrict__ BT,
    const float* __restrict__ bias,
    unsigned short* __restrict__ Cbf, float* __restrict__ Cf,
    int M, int N, int K)
{
    const int tid  = threadIdx.x;
    const int lane = tid & 31;
    const int wave = tid >> 5;                    // 0..3
    const int m0   = blockIdx.x * 16;
    const int n0   = blockIdx.y * 256 + wave * 64; // this wave's 64-col span
    const int hi   = (lane >> 4) & 1;
    const int lrow = lane & 15;

    __shared__ unsigned int ldsA[16 * 32];         // 16 rows x 64 bf16 (32 u32/row)
    __shared__ unsigned int ldsB[4][4][16][32];    // [wave][tile][col][k-pair]

    const float*        Af  = (const float*)Av;
    const unsigned int* Au  = (const unsigned int*)Av;
    const unsigned int* BTu = (const unsigned int*)BT;

    floatx8 acc[4];
    #pragma unroll
    for (int t = 0; t < 4; ++t)
        acc[t] = (floatx8){0.f, 0.f, 0.f, 0.f, 0.f, 0.f, 0.f, 0.f};

    for (int k0 = 0; k0 < K; k0 += 64) {
        // prefetch next B chunk for this wave's first column
        if (k0 + 64 < K)
            __builtin_prefetch(&BT[(long)n0 * K + k0 + 64], 0, 1);

        // ---- stage A tile: 16x64 bf16 = 512 u32, 4 contiguous u32 per thread
        {
            int idx4 = tid * 4;                    // 0..508
            int r = idx4 >> 5, c = idx4 & 31;      // c = k-pair base in row r
            if (ABF16) {
                const unsigned int* ap = &Au[((long)(m0 + r) * K + k0) / 2 + c];
                #pragma unroll
                for (int j = 0; j < 4; ++j) ldsA[idx4 + j] = ap[j];
            } else {
                const float* ap = &Af[(long)(m0 + r) * K + k0 + 2 * c];
                #pragma unroll
                for (int j = 0; j < 4; ++j) ldsA[idx4 + j] = pack_bf2(ap[2 * j], ap[2 * j + 1]);
            }
        }
        // ---- stage B: this wave's 4 tiles = 64 cols x 64 bf16; 2 cols per lane
        #pragma unroll
        for (int cc = 0; cc < 2; ++cc) {
            int g = lane * 2 + cc;                 // 0..63
            int t = g >> 4, c = g & 15;
            const unsigned int* src = &BTu[((long)(n0 + t * 16 + c) * K + k0) / 2];
            unsigned int* dst = &ldsB[wave][t][c][0];
            #pragma unroll
            for (int j = 0; j < 32; ++j) dst[j] = src[j];
        }
        __syncthreads();

        // ---- 2 K-sub-chunks x 4 N-tiles = 8 WMMAs
        #pragma unroll
        for (int s = 0; s < 2; ++s) {
            ABPack a;
            #pragma unroll
            for (int v = 0; v < 8; ++v) {
                int ka = s * 16 + v + (hi ? 8 : 0);   // A striping
                a.u[v] = ldsA[lrow * 32 + ka];
            }
            #pragma unroll
            for (int t = 0; t < 4; ++t) {
                ABPack bm;
                #pragma unroll
                for (int v = 0; v < 8; ++v)
                    bm.u[v] = ldsB[wave][t][lrow][s * 16 + v + (hi ? 8 : 0)]; // B striping
                acc[t] = __builtin_amdgcn_wmma_f32_16x16x32_bf16(
                    false, a.v, false, bm.v, (short)0, acc[t], false, false);
            }
        }
        __syncthreads();
    }

    // ---- epilogue: element (m0 + v + hi*8, n0 + t*16 + lrow)
    #pragma unroll
    for (int t = 0; t < 4; ++t) {
        const int col = n0 + t * 16 + lrow;
        const float bsv = bias[col];
        #pragma unroll
        for (int v = 0; v < 8; ++v) {
            int r = m0 + v + hi * 8;
            float val = acc[t][v] + bsv;
            if (EPI == 1) val = gelu_exact(val);
            if (EPI == 2) Cf[(long)r * N + col] = val;
            else          Cbf[(long)r * N + col] = f2bf_bits(val);
        }
    }
}

// ---------------- fused attention per (b, h, 16-query tile) ----------------
// scores = Qh Kh^T + (Ph Ph^T)/16 ; softmax ; O = Qh + A Vh
__global__ __launch_bounds__(128) void attn_kernel(
    const unsigned short* __restrict__ Qh, const unsigned short* __restrict__ Kh,
    const unsigned short* __restrict__ Vh, const unsigned short* __restrict__ Ph,
    float* __restrict__ O)
{
    const int qt = blockIdx.x;     // 0..31 query tile
    const int h  = blockIdx.y;     // 0..7
    const int b  = blockIdx.z;     // 0..31
    const int tid  = threadIdx.x;
    const int lane = tid & 31;
    const int wave = tid >> 5;
    const int hi   = (lane >> 4) & 1;
    const int lrow = lane & 15;
    const int q0   = qt * 16;

    __shared__ float        sc[16 * 512];      // 32 KB score/prob matrix
    __shared__ unsigned int ldsQ[256];
    __shared__ unsigned int ldsP[256];
    __shared__ unsigned int ldsB[4 * 256];

    const unsigned int* Qu = (const unsigned int*)Qh;
    const unsigned int* Ku = (const unsigned int*)Kh;
    const unsigned int* Pu = (const unsigned int*)Ph;

    // stage the 16x32 Qh and Ph tiles (u32 row stride = DD/2 = 128; head offset h*16)
    for (int idx = tid; idx < 256; idx += 128) {
        int r = idx >> 4, c = idx & 15;
        long base = (long)(b * SS + q0 + r) * 128 + h * 16;
        ldsQ[idx] = Qu[base + c];
        ldsP[idx] = Pu[base + c];
    }
    __syncthreads();

    ABPack aQ, aP;
    #pragma unroll
    for (int v = 0; v < 8; ++v) {
        int ka = v + (hi ? 8 : 0);
        aQ.u[v] = ldsQ[lrow * 16 + ka];
        aP.u[v] = ldsP[lrow * 16 + ka];
    }

    const floatx8 z = {0.f, 0.f, 0.f, 0.f, 0.f, 0.f, 0.f, 0.f};

    // ---- phase 1: scores. wave w owns key tiles w*8 .. w*8+7 ----
    for (int j = 0; j < 8; ++j) {
        int k0 = (wave * 8 + j) * 16;
        // stage Kh^T B tile: col = key, K-run over d (contiguous)
        #pragma unroll
        for (int t = 0; t < 8; ++t) {
            int idx = lane * 8 + t;
            int c = idx >> 4, off = idx & 15;
            ldsB[wave * 256 + idx] = Ku[(long)(b * SS + k0 + c) * 128 + h * 16 + off];
        }
        ABPack bK;
        #pragma unroll
        for (int v = 0; v < 8; ++v)
            bK.u[v] = ldsB[wave * 256 + lrow * 16 + v + (hi ? 8 : 0)];
        floatx8 cQ = __builtin_amdgcn_wmma_f32_16x16x32_bf16(false, aQ.v, false, bK.v,
                                                             (short)0, z, false, false);
        // restage with Ph keys (per-wave LDS ops are in order)
        #pragma unroll
        for (int t = 0; t < 8; ++t) {
            int idx = lane * 8 + t;
            int c = idx >> 4, off = idx & 15;
            ldsB[wave * 256 + idx] = Pu[(long)(b * SS + k0 + c) * 128 + h * 16 + off];
        }
        ABPack bP;
        #pragma unroll
        for (int v = 0; v < 8; ++v)
            bP.u[v] = ldsB[wave * 256 + lrow * 16 + v + (hi ? 8 : 0)];
        floatx8 cP = __builtin_amdgcn_wmma_f32_16x16x32_bf16(false, aP.v, false, bP.v,
                                                             (short)0, z, false, false);
        #pragma unroll
        for (int v = 0; v < 8; ++v)
            sc[(v + hi * 8) * 512 + k0 + lrow] = cQ[v] + 0.0625f * cP[v];
    }
    __syncthreads();

    // ---- phase 2: softmax, 8 lanes per row ----
    {
        int r = tid >> 3, sub = tid & 7;
        float mx = -3.0e38f;
        for (int i = 0; i < 64; ++i)
            mx = fmaxf(mx, sc[r * 512 + sub + i * 8]);
        for (int m = 1; m < 8; m <<= 1) mx = fmaxf(mx, __shfl_xor(mx, m, 8));
        float sum = 0.f;
        for (int i = 0; i < 64; ++i) {
            float e = __expf(sc[r * 512 + sub + i * 8] - mx);
            sc[r * 512 + sub + i * 8] = e;
            sum += e;
        }
        for (int m = 1; m < 8; m <<= 1) sum += __shfl_xor(sum, m, 8);
        float inv = 1.0f / sum;
        for (int i = 0; i < 64; ++i)
            sc[r * 512 + sub + i * 8] *= inv;
    }
    __syncthreads();

    // ---- phase 3: O tile = A(16x512) @ Vh(512x32) + Qh. waves 0,1 own d-tiles ----
    if (wave < 2) {
        const int nt = wave;                       // d tile: nt*16 .. nt*16+15
        floatx8 acc = z;
        for (int kc = 0; kc < 16; ++kc) {
            int kb = kc * 32;
            // A regs: probabilities -> bf16 pairs in A striping
            ABPack a;
            #pragma unroll
            for (int v = 0; v < 8; ++v) {
                int k = kb + 2 * v + (hi ? 16 : 0);
                a.u[v] = pack_bf2(sc[lrow * 512 + k], sc[lrow * 512 + k + 1]);
            }
            // B regs: Vh chunk [32 keys][16 d], col = d, K-run over keys (strided)
            #pragma unroll
            for (int t = 0; t < 8; ++t) {
                int idx = lane * 8 + t;
                int c = idx >> 4, off = idx & 15;
                int key = kb + off * 2;
                const unsigned short* vp =
                    Vh + (long)(b * SS + key) * DD + h * DHD + nt * 16 + c;
                unsigned int lo = vp[0], hi2 = vp[DD];
                ldsB[wave * 256 + idx] = lo | (hi2 << 16);
            }
            ABPack bV;
            #pragma unroll
            for (int v = 0; v < 8; ++v)
                bV.u[v] = ldsB[wave * 256 + lrow * 16 + v + (hi ? 8 : 0)];
            acc = __builtin_amdgcn_wmma_f32_16x16x32_bf16(false, a.v, false, bV.v,
                                                          (short)0, acc, false, false);
        }
        // epilogue: O = acc + Qh
        int d = h * DHD + nt * 16 + lrow;
        #pragma unroll
        for (int v = 0; v < 8; ++v) {
            int q = q0 + v + hi * 8;
            float qv = bf2f(Qh[(long)(b * SS + q) * DD + d]);
            O[(long)(b * SS + q) * DD + d] = acc[v] + qv;
        }
    }
}

// ---------------- LayerNorm over D=256, one wave32 per row ----------------
// Y = LN(X (+ Xadd)) * g + beta ; optional bf16 mirror in Ybf
__global__ __launch_bounds__(32) void ln_kernel(
    const float* __restrict__ X, const float* __restrict__ Xadd,
    const float* __restrict__ g, const float* __restrict__ be,
    float* __restrict__ Y, unsigned short* __restrict__ Ybf)
{
    const int r    = blockIdx.x;
    const int lane = threadIdx.x;
    float x[8];
    float s = 0.f;
    #pragma unroll
    for (int i = 0; i < 8; ++i) {
        int c = lane + 32 * i;
        float v = X[(long)r * DD + c];
        if (Xadd) v += Xadd[(long)r * DD + c];
        x[i] = v; s += v;
    }
    for (int m = 1; m < 32; m <<= 1) s += __shfl_xor(s, m, 32);
    float mu = s * (1.0f / DD);
    float vs = 0.f;
    #pragma unroll
    for (int i = 0; i < 8; ++i) { float d = x[i] - mu; vs += d * d; }
    for (int m = 1; m < 32; m <<= 1) vs += __shfl_xor(vs, m, 32);
    float rstd = rsqrtf(vs * (1.0f / DD) + 1e-5f);
    #pragma unroll
    for (int i = 0; i < 8; ++i) {
        int c = lane + 32 * i;
        float y = (x[i] - mu) * rstd * g[c] + be[c];
        Y[(long)r * DD + c] = y;
        if (Ybf) Ybf[(long)r * DD + c] = f2bf_bits(y);
    }
}

// ---------------- host-side orchestration ----------------
extern "C" void kernel_launch(void* const* d_in, const int* in_sizes, int n_in,
                              void* d_out, int out_size, void* d_ws, size_t ws_size,
                              hipStream_t stream) {
    const float* Q     = (const float*)d_in[0];
    const float* Kin   = (const float*)d_in[1];
    const float* p     = (const float*)d_in[2];
    const float* Wq    = (const float*)d_in[3];
    const float* bq    = (const float*)d_in[4];
    const float* Wk    = (const float*)d_in[5];
    const float* bk    = (const float*)d_in[6];
    const float* Wv    = (const float*)d_in[7];
    const float* bv    = (const float*)d_in[8];
    const float* Wp    = (const float*)d_in[9];
    const float* bp    = (const float*)d_in[10];
    const float* g0    = (const float*)d_in[11];
    const float* beta0 = (const float*)d_in[12];
    const float* W1    = (const float*)d_in[13];
    const float* b1    = (const float*)d_in[14];
    const float* W2    = (const float*)d_in[15];
    const float* b2    = (const float*)d_in[16];
    const float* g1    = (const float*)d_in[17];
    const float* beta1 = (const float*)d_in[18];
    float* out = (float*)d_out;

    // workspace layout (all sizes are multiples of 256 B)
    char* w = (char*)d_ws;
    unsigned short* WqT = (unsigned short*)w;                       w += (long)DD * DD * 2;
    unsigned short* WkT = (unsigned short*)w;                       w += (long)DD * DD * 2;
    unsigned short* WvT = (unsigned short*)w;                       w += (long)DD * DD * 2;
    unsigned short* W1T = (unsigned short*)w;                       w += (long)DD * DFF * 2;   // [DFF][DD]
    unsigned short* W2T = (unsigned short*)w;                       w += (long)DFF * DD * 2;   // [DD][DFF]
    unsigned short* Qh  = (unsigned short*)w;                       w += (long)NROW * DD * 2;
    unsigned short* Kh  = (unsigned short*)w;                       w += (long)NROW * DD * 2;
    unsigned short* Vh  = (unsigned short*)w;                       w += (long)NROW * DD * 2;
    unsigned short* Ph  = (unsigned short*)w;                       w += (long)NROW * DD * 2;
    float*          O   = (float*)w;                                w += (long)NROW * DD * 4;
    float*          Oln = (float*)w;                                w += (long)NROW * DD * 4;
    unsigned short* Olb = (unsigned short*)w;                       w += (long)NROW * DD * 2;
    unsigned short* Hbf = (unsigned short*)w;                       w += (long)NROW * DFF * 2;
    float*          ff  = (float*)w;                                w += (long)NROW * DD * 4;
    (void)ws_size; (void)in_sizes; (void)n_in; (void)out_size;

    // 1) weight convert+transpose
    tconv_kernel<<<dim3((DD * DD + 255) / 256), 256, 0, stream>>>(Wq, WqT, DD, DD);
    tconv_kernel<<<dim3((DD * DD + 255) / 256), 256, 0, stream>>>(Wk, WkT, DD, DD);
    tconv_kernel<<<dim3((DD * DD + 255) / 256), 256, 0, stream>>>(Wv, WvT, DD, DD);
    tconv_kernel<<<dim3((DD * DFF + 255) / 256), 256, 0, stream>>>(W1, W1T, DD, DFF);
    tconv_kernel<<<dim3((DFF * DD + 255) / 256), 256, 0, stream>>>(W2, W2T, DFF, DD);

    // 2) positional projection
    pe_kernel<<<dim3(NROW), 256, 0, stream>>>(p, Wp, bp, Ph);

    // 3) Q/K/V projections (fp32 A, WMMA bf16, bf16 out); block span = 16 x 256
    dim3 gproj(NROW / 16, DD / 256);
    gemm_kernel<false, 0><<<gproj, 128, 0, stream>>>(Q,   WqT, bq, Qh, nullptr, NROW, DD, DD);
    gemm_kernel<false, 0><<<gproj, 128, 0, stream>>>(Kin, WkT, bk, Kh, nullptr, NROW, DD, DD);
    gemm_kernel<false, 0><<<gproj, 128, 0, stream>>>(Kin, WvT, bv, Vh, nullptr, NROW, DD, DD);

    // 4) fused attention
    attn_kernel<<<dim3(SS / 16, HH, BB), 128, 0, stream>>>(Qh, Kh, Vh, Ph, O);

    // 5) LayerNorm 0 (fp32 + bf16 mirror for FFN)
    ln_kernel<<<dim3(NROW), 32, 0, stream>>>(O, nullptr, g0, beta0, Oln, Olb);

    // 6) FFN1: gelu(Oln @ W1 + b1) -> bf16
    gemm_kernel<true, 1><<<dim3(NROW / 16, DFF / 256), 128, 0, stream>>>(
        Olb, W1T, b1, Hbf, nullptr, NROW, DFF, DD);

    // 7) FFN2: Hbf @ W2 + b2 -> fp32
    gemm_kernel<true, 2><<<dim3(NROW / 16, DD / 256), 128, 0, stream>>>(
        Hbf, W2T, b2, nullptr, ff, NROW, DD, DFF);

    // 8) LayerNorm 1 with residual -> d_out
    ln_kernel<<<dim3(NROW), 32, 0, stream>>>(Oln, ff, g1, beta1, out, nullptr);
}
